// TtWhisperAttention_30958124269590
// MI455X (gfx1250) — matrix operand order repaired and checked by reading, not compiled, not run on hardware
//
#include <hip/hip_runtime.h>

// ---------------- problem constants ----------------
#define E     1024
#define BATCH 4
#define SEQ   1500
#define SPAD  1504          // SEQ padded to multiple of 32
#define MROWS 6000          // BATCH*SEQ
#define MPAD  6016          // MROWS padded to multiple of 64
#define HEADS 16
#define BH    64            // BATCH*HEADS
#define DH    64
#define NCHUNK (SPAD / 32)  // 47 key chunks
#define QTILES (SPAD / 16)  // 94 q-tiles per bh

// flash kernel dynamic-LDS layout (byte offsets from LDS base)
#define KS_OFF 0            // 2 x 4KB K chunk (32 keys x 64 d, bf16)
#define VS_OFF 8192         // 2 x 4KB V chunk (64 d x 32 keys, bf16)
#define AB_OFF 16384        // 8 waves x 64B alpha/l broadcast
#define LDSBYTES 16896

typedef __bf16 v4bf  __attribute__((ext_vector_type(4)));
typedef __bf16 v8bf  __attribute__((ext_vector_type(8)));
typedef __bf16 v16bf __attribute__((ext_vector_type(16)));
typedef float  v4f   __attribute__((ext_vector_type(4)));
typedef float  v8f   __attribute__((ext_vector_type(8)));

// ---------------- WMMA fragment helpers ----------------
__device__ __forceinline__ v16bf bfcat(v8bf lo, v8bf hi) {
  return __builtin_shufflevector(lo, hi, 0,1,2,3,4,5,6,7,8,9,10,11,12,13,14,15);
}
// A fragment (16x32 bf16). Lane holds row M=lane&15; lanes 0-15 get K chunks
// {0..7,16..23}, lanes 16-31 get {8..15,24..31}. Caller passes
// row_ptr + k0 + 8*(lane>>4); chunks are at +0 and +16 elements.
__device__ __forceinline__ v16bf ldA(const __bf16* p) {
  v8bf lo = *(const v8bf*)p;
  v8bf hi = *(const v8bf*)(p + 16);
  return bfcat(lo, hi);
}
// B fragment (32x16 bf16). Lane holds column N=lane&15 with 16 contiguous K
// values starting at 16*(lane>>4).
__device__ __forceinline__ v16bf ldB(const __bf16* p) {
  v8bf lo = *(const v8bf*)p;
  v8bf hi = *(const v8bf*)(p + 8);
  return bfcat(lo, hi);
}
__device__ __forceinline__ v8f wmma_bf16(v16bf a, v16bf b, v8f c) {
  return __builtin_amdgcn_wmma_f32_16x16x32_bf16(false, a, false, b, (short)0, c,
                                                 false, false);
}
// raw v_exp_f32: args are always <= 0, underflow-to-zero is wanted semantics
__device__ __forceinline__ float fast_exp2(float x) {
  return __builtin_amdgcn_exp2f(x);
}
// xor-16 lane exchange as one immediate ds_swizzle (SWAPX16)
__device__ __forceinline__ float xor16(float x) {
  int y = __builtin_amdgcn_ds_swizzle(__builtin_bit_cast(int, x), 0x401F);
  return __builtin_bit_cast(float, y);
}

// ---------------- CDNA5 async global->LDS copy (ASYNCcnt-tracked) ----------
__device__ __forceinline__ void async_b128(unsigned lds_off, const void* gptr) {
  asm volatile("global_load_async_to_lds_b128 %0, %1, off"
               :: "v"(lds_off), "v"(gptr) : "memory");
}
__device__ __forceinline__ void wait_async0() {
  asm volatile("s_wait_asynccnt 0x0" ::: "memory");
}

// ---------------- fp32 -> bf16 convert (x4) with zero padding ---------------
__global__ void cvt_pad_kernel(const float* __restrict__ src, __bf16* __restrict__ dst,
                               long n4src, long n4dst) {
  long i = (long)blockIdx.x * blockDim.x + threadIdx.x;   // 4-element groups
  if (i < n4dst) {
    v4f v = {};
    if (i < n4src) v = *(const v4f*)(src + 4 * i);
    *(v4bf*)(dst + 4 * i) = __builtin_convertvector(v, v4bf);
  }
}

// zero the 4 padded seq rows (1500..1503) of Q, K and the padded cols of Vt
__global__ void pad_tail_kernel(__bf16* __restrict__ Qb, __bf16* __restrict__ Kb,
                                __bf16* __restrict__ Vt) {
  int i = blockIdx.x * blockDim.x + threadIdx.x;   // 3 * 64 * 4 * 64 threads
  int t   = i >> 14;
  int rem = i & 16383;
  int bh  = rem >> 8;
  int rr  = rem & 255;
  int r   = rr >> 6, d = rr & 63;
  int s   = SEQ + r;
  if (t == 0)      Qb[((size_t)bh * SPAD + s) * DH + d] = (__bf16)0.0f;
  else if (t == 1) Kb[((size_t)bh * SPAD + s) * DH + d] = (__bf16)0.0f;
  else             Vt[((size_t)bh * DH + d) * SPAD + s] = (__bf16)0.0f;
}

// ---------------- shared GEMM inner loop: 64x64 wave tile, y = x @ W^T ------
// ~32 FLOP per byte loaded per 64x64 wave tile -> no LDS staging needed for the GEMMs.
__device__ __forceinline__ void gemm_tile_acc(const __bf16* __restrict__ A,
                                              const __bf16* __restrict__ W,
                                              int m0, int n0, v8f acc[4][4]) {
  const int lane = threadIdx.x & 31;
  const int l15 = lane & 15, hi = lane >> 4;
  const __bf16* ap[4];
#pragma unroll
  for (int mt = 0; mt < 4; mt++)
    ap[mt] = A + (size_t)(m0 + mt * 16 + l15) * E + 8 * hi;
  const __bf16* wp = W + (size_t)(n0 + l15) * E + 8 * hi;
  for (int k0 = 0; k0 < E; k0 += 32) {
    v16bf av[4];
#pragma unroll
    for (int mt = 0; mt < 4; mt++) av[mt] = ldA(ap[mt] + k0);
#pragma unroll
    for (int nt = 0; nt < 4; nt++) {
      v16bf bw = ldA(wp + (size_t)(nt * 16) * E + k0);
#pragma unroll
      for (int mt = 0; mt < 4; mt++)
        acc[mt][nt] = wmma_bf16(av[mt], bw, acc[mt][nt]);
    }
  }
}

// ---------------- QKV projection (bf16 out, head-split or transposed) -------
template <bool HAS_BIAS, bool VTRANS>
__global__ __launch_bounds__(256) void proj_kernel(const __bf16* __restrict__ X,
                                                   const __bf16* __restrict__ W,
                                                   const float* __restrict__ bias,
                                                   __bf16* __restrict__ out,
                                                   float scale) {
  const int wave = threadIdx.x >> 5;
  const int wg = blockIdx.x * 8 + wave;           // 94 * 16 wave tiles
  const int m0 = (wg >> 4) * 64;
  const int n0 = (wg & 15) * 64;
  v8f acc[4][4] = {};
  gemm_tile_acc(X, W, m0, n0, acc);

  const int lane = threadIdx.x & 31;
  const int l15 = lane & 15, hi = lane >> 4;
#pragma unroll
  for (int nt = 0; nt < 4; nt++) {
    const int n = n0 + nt * 16 + l15;
    const float bv = HAS_BIAS ? bias[n] : 0.0f;
    const int h = n >> 6, d = n & 63;
#pragma unroll
    for (int mt = 0; mt < 4; mt++) {
#pragma unroll
      for (int r = 0; r < 8; r++) {
        int m = m0 + mt * 16 + r + 8 * hi;        // C layout: M = r + 8*(lane>=16)
        if (m < MROWS) {
          int b = m / SEQ;
          int s = m - b * SEQ;
          float v = (acc[mt][nt][r] + bv) * scale;
          if (VTRANS)   // V stored [bh][DH][SPAD]: P@V B-frags are contiguous-K
            out[(((size_t)(b * HEADS + h) * DH) + d) * SPAD + s] = (__bf16)v;
          else          // Q/K stored [bh][SPAD][DH]
            out[(((size_t)(b * HEADS + h) * SPAD) + s) * DH + d] = (__bf16)v;
        }
      }
    }
  }
}

// ---------------- flash attention -------------------------------------------
// Block = 8 waves of the SAME bh; K/V chunks double-buffered into LDS with
// global_load_async_to_lds_b128 (ASYNCcnt) -> 8x less L2 traffic.
// Scores computed transposed (S^T = K @ Q^T): softmax reductions are in-lane
// + one ds_swizzle, AND the C-layout of S^T coincides exactly with the
// A-fragment layout of P for P@V, so P never touches LDS (pure register cvt).
__device__ __forceinline__ void stage_chunk(const __bf16* __restrict__ K,
                                            const __bf16* __restrict__ Vt,
                                            int bh, int k0, int buf, int t) {
  // K chunk: 32 rows x 64 d bf16 = 4KB, fully contiguous
  const char* kg = (const char*)(K + ((size_t)bh * SPAD + k0) * DH) + t * 16;
  async_b128((unsigned)(KS_OFF + buf * 4096 + t * 16), kg);
  // V chunk: 64 rows (d) x 32 keys bf16 = 4KB; 4 x 16B segments per row
  const char* vg = (const char*)(Vt + ((size_t)(bh * DH + (t >> 2)) * SPAD) +
                                 k0 + (t & 3) * 8);
  async_b128((unsigned)(VS_OFF + buf * 4096 + t * 16), vg);
}

__global__ __launch_bounds__(256) void flash_kernel(const __bf16* __restrict__ Q,
                                                    const __bf16* __restrict__ K,
                                                    const __bf16* __restrict__ Vt,
                                                    __bf16* __restrict__ attn) {
  extern __shared__ char smem[];
  const int t = threadIdx.x;
  const int lane = t & 31, wave = t >> 5;
  const int bh = blockIdx.x / 12;
  const int qblk = blockIdx.x % 12;
  const int qt_raw = qblk * 8 + wave;                  // 0..95
  const int qt = qt_raw < QTILES ? qt_raw : QTILES - 1;
  const int b = bh >> 4, h = bh & 15;
  const int l15 = lane & 15, hi = lane >> 4;

  // Q as B-fragments (lane = query column, table-order d chunks like ldA);
  // scaling * log2(e) already baked in by the projection.
  const __bf16* qrow = Q + ((size_t)bh * SPAD + qt * 16 + l15) * DH;
  v16bf qf0 = ldA(qrow + 8 * hi);
  v16bf qf1 = ldA(qrow + 32 + 8 * hi);

  v8f o0 = {}, o1 = {}, o2 = {}, o3 = {};
  float mst = -1e30f, lst = 0.0f;     // per-query running max / sum (scalars)

  float* aw = (float*)(smem + AB_OFF + wave * 64);

  stage_chunk(K, Vt, bh, 0, 0, t);    // prime buffer 0

  for (int c = 0; c < NCHUNK; ++c) {
    wait_async0();                    // my slice of buffer c&1 has landed
    __syncthreads();                  // everyone's slice landed; prior chunk
                                      // fully consumed -> safe to overwrite
    if (c + 1 < NCHUNK) stage_chunk(K, Vt, bh, (c + 1) * 32, (c + 1) & 1, t);

    const __bf16* ks = (const __bf16*)(smem + KS_OFF + (c & 1) * 4096);
    const __bf16* vs = (const __bf16*)(smem + VS_OFF + (c & 1) * 4096);
    const int k0 = c * 32;

    // all K + V fragments up-front: one DS clause, V loads overlap softmax
    v16bf ka0 = ldA(ks + l15 * DH + 8 * hi);
    v16bf ka1 = ldA(ks + l15 * DH + 32 + 8 * hi);
    v16bf ka2 = ldA(ks + (16 + l15) * DH + 8 * hi);
    v16bf ka3 = ldA(ks + (16 + l15) * DH + 32 + 8 * hi);
    v16bf vb0 = ldA(vs + (l15) * 32 + 8 * hi);
    v16bf vb1 = ldA(vs + (16 + l15) * 32 + 8 * hi);
    v16bf vb2 = ldA(vs + (32 + l15) * 32 + 8 * hi);
    v16bf vb3 = ldA(vs + (48 + l15) * 32 + 8 * hi);

    // S^T tiles: rows = keys, cols = queries
    v8f st0 = {}, st1 = {};
    st0 = wmma_bf16(ka0, qf0, st0);
    st0 = wmma_bf16(ka1, qf1, st0);
    st1 = wmma_bf16(ka2, qf0, st1);
    st1 = wmma_bf16(ka3, qf1, st1);

    if (k0 + 32 > SEQ) {              // tail: mask keys >= SEQ (key = C row)
#pragma unroll
      for (int r = 0; r < 8; r++) {
        if (k0 + r + 8 * hi >= SEQ)      st0[r] = -1e30f;
        if (k0 + 16 + r + 8 * hi >= SEQ) st1[r] = -1e30f;
      }
    }
    // online softmax: 16 in-register key scores per lane + xor-16 partner
    float rm = st0[0];
#pragma unroll
    for (int r = 1; r < 8; r++) rm = fmaxf(rm, st0[r]);
#pragma unroll
    for (int r = 0; r < 8; r++) rm = fmaxf(rm, st1[r]);
    rm = fmaxf(rm, xor16(rm));
    float mn = fmaxf(mst, rm);
    float al = fast_exp2(mst - mn);
    float rs = 0.0f;
#pragma unroll
    for (int r = 0; r < 8; r++) {
      st0[r] = fast_exp2(st0[r] - mn);
      st1[r] = fast_exp2(st1[r] - mn);
      rs += st0[r] + st1[r];
    }
    rs += xor16(rs);
    lst = lst * al + rs;
    mst = mn;

    // P's A-fragment == C-layout of S^T: pure register conversion, no LDS
    v16bf pa = bfcat(__builtin_convertvector(st0, v8bf),
                     __builtin_convertvector(st1, v8bf));

    // rescale O with alpha of each O-row's query (broadcast ds_load_b128)
    aw[l15] = al;                     // both halves write identical value
    v4f a_lo = *(const v4f*)&aw[8 * hi];
    v4f a_hi = *(const v4f*)&aw[8 * hi + 4];
#pragma unroll
    for (int r = 0; r < 8; r++) {
      float a = (r < 4) ? a_lo[r] : a_hi[r - 4];
      o0[r] *= a; o1[r] *= a; o2[r] *= a; o3[r] *= a;
    }
    // O += P @ V from the staged V chunk (row = d, 32 contiguous keys)
    o0 = wmma_bf16(pa, vb0, o0);
    o1 = wmma_bf16(pa, vb1, o1);
    o2 = wmma_bf16(pa, vb2, o2);
    o3 = wmma_bf16(pa, vb3, o3);
  }
  // broadcast l per query, normalize, merge heads
  aw[l15] = lst;
  v4f l_lo = *(const v4f*)&aw[8 * hi];
  v4f l_hi = *(const v4f*)&aw[8 * hi + 4];
#pragma unroll
  for (int r = 0; r < 8; r++) {
    int s = qt * 16 + r + 8 * hi;
    if (qt_raw < QTILES && s < SEQ) {
      float lv = (r < 4) ? l_lo[r] : l_hi[r - 4];
      float inv = 1.0f / lv;
      __bf16* orow = attn + ((size_t)b * SEQ + s) * E + h * DH;
      orow[l15]      = (__bf16)(o0[r] * inv);
      orow[16 + l15] = (__bf16)(o1[r] * inv);
      orow[32 + l15] = (__bf16)(o2[r] * inv);
      orow[48 + l15] = (__bf16)(o3[r] * inv);
    }
  }
}

// ---------------- output projection: fp32 result + bias ----------------
__global__ __launch_bounds__(256) void outproj_kernel(const __bf16* __restrict__ A,
                                                      const __bf16* __restrict__ W,
                                                      const float* __restrict__ bias,
                                                      float* __restrict__ out) {
  const int wave = threadIdx.x >> 5;
  const int wg = blockIdx.x * 8 + wave;
  const int m0 = (wg >> 4) * 64;
  const int n0 = (wg & 15) * 64;
  v8f acc[4][4] = {};
  gemm_tile_acc(A, W, m0, n0, acc);

  const int lane = threadIdx.x & 31;
  const int l15 = lane & 15, hi = lane >> 4;
#pragma unroll
  for (int nt = 0; nt < 4; nt++) {
    const int n = n0 + nt * 16 + l15;
    const float bv = bias[n];
#pragma unroll
    for (int mt = 0; mt < 4; mt++) {
#pragma unroll
      for (int r = 0; r < 8; r++) {
        int m = m0 + mt * 16 + r + 8 * hi;
        if (m < MROWS) out[(size_t)m * E + n] = acc[mt][nt][r] + bv;
      }
    }
  }
}

// ---------------- host-side orchestration ----------------
extern "C" void kernel_launch(void* const* d_in, const int* in_sizes, int n_in,
                              void* d_out, int out_size, void* d_ws, size_t ws_size,
                              hipStream_t stream) {
  (void)in_sizes; (void)n_in; (void)out_size; (void)ws_size;
  const float* hs  = (const float*)d_in[0];
  const float* q_w = (const float*)d_in[1];
  const float* q_b = (const float*)d_in[2];
  const float* k_w = (const float*)d_in[3];
  const float* v_w = (const float*)d_in[4];
  const float* v_b = (const float*)d_in[5];
  const float* o_w = (const float*)d_in[6];
  const float* o_b = (const float*)d_in[7];
  float* out = (float*)d_out;

  char* ws = (char*)d_ws;
  size_t off = 0;
  auto take = [&](size_t bytes) {
    char* p = ws + off;
    off += (bytes + 255) & ~(size_t)255;
    return p;
  };
  __bf16* Xb   = (__bf16*)take((size_t)MPAD * E * 2);
  __bf16* Wq   = (__bf16*)take((size_t)E * E * 2);
  __bf16* Wk   = (__bf16*)take((size_t)E * E * 2);
  __bf16* Wv   = (__bf16*)take((size_t)E * E * 2);
  __bf16* Wo   = (__bf16*)take((size_t)E * E * 2);
  __bf16* Qb   = (__bf16*)take((size_t)BH * SPAD * DH * 2);
  __bf16* Kb   = (__bf16*)take((size_t)BH * SPAD * DH * 2);
  __bf16* Vt   = (__bf16*)take((size_t)BH * DH * SPAD * 2);
  __bf16* Attn = (__bf16*)take((size_t)MPAD * E * 2);

  // 1) fp32 -> bf16, 4 elements per thread (all sizes divisible by 4)
  {
    long n4src = (long)MROWS * E / 4, n4dst = (long)MPAD * E / 4;
    cvt_pad_kernel<<<(unsigned)((n4dst + 255) / 256), 256, 0, stream>>>(hs, Xb, n4src, n4dst);
    long n4w = (long)E * E / 4;
    unsigned gw = (unsigned)((n4w + 255) / 256);
    cvt_pad_kernel<<<gw, 256, 0, stream>>>(q_w, Wq, n4w, n4w);
    cvt_pad_kernel<<<gw, 256, 0, stream>>>(k_w, Wk, n4w, n4w);
    cvt_pad_kernel<<<gw, 256, 0, stream>>>(v_w, Wv, n4w, n4w);
    cvt_pad_kernel<<<gw, 256, 0, stream>>>(o_w, Wo, n4w, n4w);
  }
  // 2) zero seq-padding rows of Q/K/Vt
  pad_tail_kernel<<<192, 256, 0, stream>>>(Qb, Kb, Vt);

  // 3) projections: Q gets scaling * log2(e) baked in so softmax uses exp2
  const float qscale = 0.125f * 1.44269504088896340736f;
  dim3 gproj(188), blk(256);     // 94 m-tiles * 16 n-tiles / 8 waves per block
  proj_kernel<true,  false><<<gproj, blk, 0, stream>>>(Xb, Wq, q_b, Qb, qscale);
  proj_kernel<false, false><<<gproj, blk, 0, stream>>>(Xb, Wk, nullptr, Kb, 1.0f);
  proj_kernel<true,  true ><<<gproj, blk, 0, stream>>>(Xb, Wv, v_b, Vt, 1.0f);

  // 4) flash attention: 64 bh x 12 q-blocks, 8 waves/block share staged K/V
  flash_kernel<<<768, 256, LDSBYTES, stream>>>(Qb, Kb, Vt, Attn);

  // 5) output projection -> fp32 d_out
  outproj_kernel<<<gproj, blk, 0, stream>>>(Attn, Wo, o_b, out);
}
